// Decoder_33002528702695
// MI455X (gfx1250) — matrix operand order repaired and checked by reading, not compiled, not run on hardware
//
#include <hip/hip_runtime.h>
#include <hip/hip_bf16.h>
#include <stdint.h>

typedef unsigned short u16;
typedef __attribute__((ext_vector_type(16))) __bf16 v16bf;
typedef __attribute__((ext_vector_type(8)))  float  v8f;

#define N_MEL 80
#define ENC   512
#define ARNN  1024
#define DRNN  1024
#define PRE   256
#define ADIM  128
#define NFILT 32
#define KS    31
#define BSZ   16
#define TENC  128
#define TDEC  128
#define GA    4096          // 4*ARNN == 4*DRNN
#define KA    (PRE+ENC+ARNN)    // 1792 : attention cat(x, ctx, h)
#define KD    (ARNN+ENC+DRNN)   // 2560 : decoder  cat(attn_h, ctx, dec_h)

struct __attribute__((aligned(16))) U128 { unsigned int x, y, z, w; };
struct __attribute__((aligned(16))) B256 { U128 lo, hi; };
struct __attribute__((aligned(16))) F4   { float x, y, z, w; };

__device__ __forceinline__ u16 f2bf(float f) {
  unsigned u = __builtin_bit_cast(unsigned, f);
  u += 0x7FFFu + ((u >> 16) & 1u);
  return (u16)(u >> 16);
}
__device__ __forceinline__ float sigf(float x) { return 1.0f / (1.0f + expf(-x)); }

// ================= prep kernels (once per call) =================

// pack [wih | whh] (fp32) -> bf16 row-major [GA, Ktot]
__global__ void k_pack_w(const float* __restrict__ wih, int kih,
                         const float* __restrict__ whh, int khh,
                         u16* __restrict__ out) {
  const int ktot = kih + khh;
  size_t idx = (size_t)blockIdx.x * blockDim.x + threadIdx.x;
  if (idx >= (size_t)GA * ktot) return;
  int n = (int)(idx / ktot), k = (int)(idx % ktot);
  float v = (k < kih) ? wih[(size_t)n * kih + k] : whh[(size_t)n * khh + (k - kih)];
  out[idx] = f2bf(v);
}

// prenet layer1: hid1[r,j] = relu(sum_m dec_in[r,m] * w1[j,m]), r = t*16+b
__global__ void k_prenet1(const float* __restrict__ mels, const float* __restrict__ w1,
                          float* __restrict__ hid1) {
  int idx = blockIdx.x * blockDim.x + threadIdx.x;  // 2048*256
  if (idx >= TDEC * BSZ * PRE) return;
  int r = idx / PRE, j = idx % PRE;
  int t = r / BSZ, b = r % BSZ;
  float acc = 0.f;
  if (t > 0) {
    const float* mr = mels + (size_t)b * N_MEL * TDEC + (t - 1);
    const float* wr = w1 + (size_t)j * N_MEL;
    for (int m = 0; m < N_MEL; ++m) acc += mr[m * TDEC] * wr[m];
  }
  hid1[idx] = fmaxf(acc, 0.f);
}
// prenet layer2 -> bf16 activations for all steps
__global__ void k_prenet2(const float* __restrict__ hid1, const float* __restrict__ w2,
                          u16* __restrict__ x_all) {
  int idx = blockIdx.x * blockDim.x + threadIdx.x;
  if (idx >= TDEC * BSZ * PRE) return;
  int r = idx / PRE, j = idx % PRE;
  const float* hr = hid1 + (size_t)r * PRE;
  const float* wr = w2 + (size_t)j * PRE;
  float acc = 0.f;
  for (int k = 0; k < PRE; ++k) acc += hr[k] * wr[k];
  x_all[idx] = f2bf(fmaxf(acc, 0.f));
}

// proc_mem[b,t,a] = memory[b,t,:] . memory_w[a,:]
__global__ void k_procmem(const float* __restrict__ memory, const float* __restrict__ mw,
                          float* __restrict__ pm) {
  int idx = blockIdx.x * blockDim.x + threadIdx.x;  // 16*128*128
  if (idx >= BSZ * TENC * ADIM) return;
  int a = idx & (ADIM - 1);
  int bt = idx >> 7;
  const float* mr = memory + (size_t)bt * ENC;
  const float* wr = mw + (size_t)a * ENC;
  float acc = 0.f;
  for (int e = 0; e < ENC; ++e) acc += mr[e] * wr[e];
  pm[idx] = acc;
}

__global__ void k_zero(unsigned int* __restrict__ p, int n_words) {
  int i = blockIdx.x * blockDim.x + threadIdx.x;
  if (i < n_words) p[i] = 0u;
}

// ================= per-step kernels =================

// gates[16,4096] = A[16,L0+L1+L2] x Wcat[4096,Ktot]^T via bf16 WMMA.
// A is three row-major segments with compile-time lengths; all per-lane base
// addresses hoisted, inner loops are pure {2+2 b128 loads, wmma, ptr bump}.
template <int L0, int L1, int L2>
__global__ void k_gates_t(const u16* __restrict__ Wcat,
                          const u16* __restrict__ seg0,
                          const u16* __restrict__ seg1,
                          const u16* __restrict__ seg2,
                          float* __restrict__ gates) {
  constexpr int KTOT = L0 + L1 + L2;
  const int wave = (blockIdx.x * blockDim.x + threadIdx.x) >> 5;  // 256 waves
  const int lane = threadIdx.x & 31;
  const int n0 = wave * 16;
  const int row = lane & 15;
  const bool lo = (lane < 16);

  // B operand: column n = W row (n0 + lane%16); K contiguous, split by lane half.
  const u16* pb = Wcat + (size_t)(n0 + row) * KTOT + (lo ? 0 : 16);
  v8f acc = {};

#define GEMM_SEG(SEG, LEN)                                                    \
  {                                                                           \
    const u16* pa = (SEG) + (size_t)row * (LEN) + (lo ? 0 : 16);              \
    _Pragma("unroll 8")                                                       \
    for (int k = 0; k < (LEN); k += 32) {                                     \
      B256 fa, fb;                                                            \
      fa.lo = *reinterpret_cast<const U128*>(pa);                             \
      fa.hi = *reinterpret_cast<const U128*>(pa + 8);                         \
      fb.lo = *reinterpret_cast<const U128*>(pb);                             \
      fb.hi = *reinterpret_cast<const U128*>(pb + 8);                         \
      acc = __builtin_amdgcn_wmma_f32_16x16x32_bf16(                          \
          false, __builtin_bit_cast(v16bf, fa),                               \
          false, __builtin_bit_cast(v16bf, fb), (short)0, acc, false, false); \
      pa += 32; pb += 32;                                                     \
    }                                                                         \
  }

  GEMM_SEG(seg0, L0)
  GEMM_SEG(seg1, L1)
  GEMM_SEG(seg2, L2)
#undef GEMM_SEG

  const int nc = n0 + row;
  const int mb = lo ? 0 : 8;
#pragma unroll
  for (int r = 0; r < 8; ++r)
    gates[(size_t)(mb + r) * GA + nc] = acc[r];
}

// pointwise LSTM cell: gate order i,f,g,o
__global__ void k_lstm(const float* __restrict__ gates,
                       const float* __restrict__ bih, const float* __restrict__ bhh,
                       float* __restrict__ h, float* __restrict__ c,
                       u16* __restrict__ h_bf) {
  int idx = blockIdx.x * blockDim.x + threadIdx.x;  // 16*1024
  if (idx >= BSZ * ARNN) return;
  int b = idx >> 10, u = idx & 1023;
  const float* g = gates + (size_t)b * GA;
  float gi = g[u]          + bih[u]          + bhh[u];
  float gf = g[ARNN + u]   + bih[ARNN + u]   + bhh[ARNN + u];
  float gg = g[2*ARNN + u] + bih[2*ARNN + u] + bhh[2*ARNN + u];
  float go = g[3*ARNN + u] + bih[3*ARNN + u] + bhh[3*ARNN + u];
  float cn = sigf(gf) * c[idx] + sigf(gi) * tanhf(gg);
  float hn = sigf(go) * tanhf(cn);
  c[idx] = cn; h[idx] = hn; h_bf[idx] = f2bf(hn);
}

// pq[b,a] = attn_h[b,:] . query_w[a,:]
__global__ void k_query(const float* __restrict__ h, const float* __restrict__ qw,
                        float* __restrict__ pq) {
  int b = blockIdx.x, a = threadIdx.x;
  const float* hr = h + (size_t)b * ARNN;
  const float* wr = qw + (size_t)a * ARNN;
  float acc = 0.f;
  for (int k = 0; k < ARNN; ++k) acc += hr[k] * wr[k];
  pq[b * ADIM + a] = acc;
}

// location conv + energies (one thread per (b, t_enc))
__global__ void k_energy(const float* __restrict__ aw, const float* __restrict__ awc,
                         const float* __restrict__ pq, const float* __restrict__ pm,
                         const float* __restrict__ cw, const float* __restrict__ ldw,
                         const float* __restrict__ vw, const int* __restrict__ mlen,
                         float* __restrict__ e) {
  __shared__ float s_aw[TENC], s_awc[TENC], s_pq[ADIM], s_v[ADIM];
  __shared__ float s_cw[NFILT * 2 * KS];   // 1984
  __shared__ float s_ld[ADIM * NFILT];     // 4096
  const int b = blockIdx.x, tt = threadIdx.x;
  s_aw[tt]  = aw[b * TENC + tt];
  s_awc[tt] = awc[b * TENC + tt];
  s_pq[tt]  = pq[b * ADIM + tt];
  s_v[tt]   = vw[tt];
  for (int i = tt; i < NFILT * 2 * KS; i += TENC) s_cw[i] = cw[i];
  for (int i = tt; i < ADIM * NFILT; i += TENC)   s_ld[i] = ldw[i];
  __syncthreads();

  float loc_f[NFILT];
#pragma unroll 4
  for (int f = 0; f < NFILT; ++f) {
    float acc = 0.f;
    const float* w0 = s_cw + f * 2 * KS;
    for (int k = 0; k < KS; ++k) {
      int p = tt + k - (KS - 1) / 2;
      if (p >= 0 && p < TENC) acc += s_aw[p] * w0[k] + s_awc[p] * w0[KS + k];
    }
    loc_f[f] = acc;
  }
  const float* pmr = pm + ((size_t)b * TENC + tt) * ADIM;
  float acc = 0.f;
  for (int a = 0; a < ADIM; ++a) {
    float s = s_pq[a] + pmr[a];
    const float* ld = s_ld + a * NFILT;
#pragma unroll
    for (int f = 0; f < NFILT; ++f) s += loc_f[f] * ld[f];
    acc += tanhf(s) * s_v[a];
  }
  if (tt >= mlen[b]) acc = -1e30f;
  e[b * TENC + tt] = acc;
}

// softmax over T_enc, context, cumulative weights, alignment output
__global__ void k_softctx(const float* __restrict__ e, const float* __restrict__ memory,
                          float* __restrict__ aw, float* __restrict__ awc,
                          float* __restrict__ ctx, u16* __restrict__ ctx_bf,
                          float* __restrict__ out_align, int t) {
  __shared__ float red[TENC], s_aw[TENC];
  const int b = blockIdx.x, tt = threadIdx.x;
  float ev = e[b * TENC + tt];
  red[tt] = ev; __syncthreads();
  for (int s = TENC / 2; s > 0; s >>= 1) {
    if (tt < s) red[tt] = fmaxf(red[tt], red[tt + s]);
    __syncthreads();
  }
  const float mx = red[0]; __syncthreads();
  float ex = expf(ev - mx);
  red[tt] = ex; __syncthreads();
  for (int s = TENC / 2; s > 0; s >>= 1) {
    if (tt < s) red[tt] += red[tt + s];
    __syncthreads();
  }
  const float inv = 1.0f / red[0];
  const float a2 = ex * inv;
  s_aw[tt] = a2;
  aw[b * TENC + tt] = a2;
  awc[b * TENC + tt] += a2;
  out_align[(size_t)b * TDEC * TENC + (size_t)t * TENC + tt] = a2;
  __syncthreads();
  // ctx[b, 4*tt .. 4*tt+3] = sum_t2 a2[t2] * memory[b, t2, 4tt..]
  F4 acc = {0.f, 0.f, 0.f, 0.f};
  for (int t2 = 0; t2 < TENC; ++t2) {
    const F4 m = *reinterpret_cast<const F4*>(memory + ((size_t)b * TENC + t2) * ENC + tt * 4);
    float wgt = s_aw[t2];
    acc.x += wgt * m.x; acc.y += wgt * m.y; acc.z += wgt * m.z; acc.w += wgt * m.w;
  }
  float* cr = ctx + (size_t)b * ENC + tt * 4;
  cr[0] = acc.x; cr[1] = acc.y; cr[2] = acc.z; cr[3] = acc.w;
  u16* cb = ctx_bf + (size_t)b * ENC + tt * 4;
  cb[0] = f2bf(acc.x); cb[1] = f2bf(acc.y); cb[2] = f2bf(acc.z); cb[3] = f2bf(acc.w);
}

// mel + gate projections for one step
__global__ void k_proj(const float* __restrict__ dec_h, const float* __restrict__ ctx,
                       const float* __restrict__ pw, const float* __restrict__ pb,
                       const float* __restrict__ gw, const float* __restrict__ gb,
                       float* __restrict__ out, int t) {
  const int b = blockIdx.x, m = threadIdx.x;
  const size_t gate_off = (size_t)BSZ * N_MEL * TDEC;
  const float* hr = dec_h + (size_t)b * DRNN;
  const float* cr = ctx + (size_t)b * ENC;
  if (m < N_MEL) {
    const float* wr = pw + (size_t)m * (DRNN + ENC);
    float acc = pb[m];
    for (int u = 0; u < DRNN; ++u) acc += hr[u] * wr[u];
    for (int ee = 0; ee < ENC; ++ee) acc += cr[ee] * wr[DRNN + ee];
    out[(size_t)b * N_MEL * TDEC + (size_t)m * TDEC + t] = acc;
  } else if (m == N_MEL) {
    float acc = gb[0];
    for (int u = 0; u < DRNN; ++u) acc += hr[u] * gw[u];
    for (int ee = 0; ee < ENC; ++ee) acc += cr[ee] * gw[DRNN + ee];
    out[gate_off + (size_t)b * TDEC + t] = acc;
  }
}

// ================= host orchestration =================
extern "C" void kernel_launch(void* const* d_in, const int* in_sizes, int n_in,
                              void* d_out, int out_size, void* d_ws, size_t ws_size,
                              hipStream_t stream) {
  const float* memory     = (const float*)d_in[0];
  const float* mels       = (const float*)d_in[1];
  const int*   mem_len    = (const int*)d_in[2];
  const float* prenet_w1  = (const float*)d_in[3];
  const float* prenet_w2  = (const float*)d_in[4];
  const float* arnn_wih   = (const float*)d_in[5];
  const float* arnn_whh   = (const float*)d_in[6];
  const float* arnn_bih   = (const float*)d_in[7];
  const float* arnn_bhh   = (const float*)d_in[8];
  const float* query_w    = (const float*)d_in[9];
  const float* memory_w   = (const float*)d_in[10];
  const float* v_w        = (const float*)d_in[11];
  const float* loc_conv_w = (const float*)d_in[12];
  const float* loc_dense_w= (const float*)d_in[13];
  const float* drnn_wih   = (const float*)d_in[14];
  const float* drnn_whh   = (const float*)d_in[15];
  const float* drnn_bih   = (const float*)d_in[16];
  const float* drnn_bhh   = (const float*)d_in[17];
  const float* proj_w     = (const float*)d_in[18];
  const float* proj_b     = (const float*)d_in[19];
  const float* gate_w     = (const float*)d_in[20];
  const float* gate_b     = (const float*)d_in[21];
  float* out = (float*)d_out;

  // ---- workspace layout ----
  char* w = (char*)d_ws;
  u16*   WcatA  = (u16*)w;   w += (size_t)GA * KA * sizeof(u16);      // 14.7 MB
  u16*   WcatD  = (u16*)w;   w += (size_t)GA * KD * sizeof(u16);      // 21.0 MB
  u16*   x_all  = (u16*)w;   w += (size_t)TDEC * BSZ * PRE * sizeof(u16);
  float* hid1   = (float*)w; w += (size_t)TDEC * BSZ * PRE * sizeof(float);
  float* pm     = (float*)w; w += (size_t)BSZ * TENC * ADIM * sizeof(float);
  float* gates  = (float*)w; w += (size_t)BSZ * GA * sizeof(float);
  // ---- zeroed state block (contiguous) ----
  char* state0 = w;
  float* attn_h = (float*)w; w += (size_t)BSZ * ARNN * sizeof(float);
  float* attn_c = (float*)w; w += (size_t)BSZ * ARNN * sizeof(float);
  float* dec_h  = (float*)w; w += (size_t)BSZ * DRNN * sizeof(float);
  float* dec_c  = (float*)w; w += (size_t)BSZ * DRNN * sizeof(float);
  float* ctx    = (float*)w; w += (size_t)BSZ * ENC * sizeof(float);
  float* aw     = (float*)w; w += (size_t)BSZ * TENC * sizeof(float);
  float* awc    = (float*)w; w += (size_t)BSZ * TENC * sizeof(float);
  float* e      = (float*)w; w += (size_t)BSZ * TENC * sizeof(float);
  float* pq     = (float*)w; w += (size_t)BSZ * ADIM * sizeof(float);
  u16* attnh_bf = (u16*)w;   w += (size_t)BSZ * ARNN * sizeof(u16);
  u16* dech_bf  = (u16*)w;   w += (size_t)BSZ * DRNN * sizeof(u16);
  u16* ctx_bf   = (u16*)w;   w += (size_t)BSZ * ENC * sizeof(u16);
  const int state_words = (int)(((char*)w - state0) / 4);

  // ---- per-call prep ----
  {
    size_t na = (size_t)GA * KA;
    k_pack_w<<<(int)((na + 255) / 256), 256, 0, stream>>>(arnn_wih, PRE + ENC,
                                                          arnn_whh, ARNN, WcatA);
    size_t nd = (size_t)GA * KD;
    k_pack_w<<<(int)((nd + 255) / 256), 256, 0, stream>>>(drnn_wih, ARNN + ENC,
                                                          drnn_whh, DRNN, WcatD);
    int np = TDEC * BSZ * PRE;
    k_prenet1<<<(np + 255) / 256, 256, 0, stream>>>(mels, prenet_w1, hid1);
    k_prenet2<<<(np + 255) / 256, 256, 0, stream>>>(hid1, prenet_w2, x_all);
    int npm = BSZ * TENC * ADIM;
    k_procmem<<<(npm + 255) / 256, 256, 0, stream>>>(memory, memory_w, pm);
    k_zero<<<(state_words + 255) / 256, 256, 0, stream>>>((unsigned int*)state0, state_words);
  }

  float* out_align = out + (size_t)BSZ * N_MEL * TDEC + (size_t)BSZ * TDEC;

  // ---- sequential decode: 8 kernels per step, serialized on `stream` ----
  for (int t = 0; t < TDEC; ++t) {
    const u16* x_t = x_all + (size_t)t * BSZ * PRE;
    // attention LSTM gates: A = [x_t | ctx | attn_h], K = 1792
    k_gates_t<PRE, ENC, ARNN><<<64, 128, 0, stream>>>(WcatA, x_t, ctx_bf, attnh_bf, gates);
    k_lstm<<<64, 256, 0, stream>>>(gates, arnn_bih, arnn_bhh, attn_h, attn_c, attnh_bf);
    k_query<<<BSZ, ADIM, 0, stream>>>(attn_h, query_w, pq);
    k_energy<<<BSZ, TENC, 0, stream>>>(aw, awc, pq, pm, loc_conv_w, loc_dense_w,
                                       v_w, mem_len, e);
    k_softctx<<<BSZ, TENC, 0, stream>>>(e, memory, aw, awc, ctx, ctx_bf, out_align, t);
    // decoder LSTM gates: A = [attn_h | ctx | dec_h], K = 2560
    k_gates_t<ARNN, ENC, DRNN><<<64, 128, 0, stream>>>(WcatD, attnh_bf, ctx_bf, dech_bf, gates);
    k_lstm<<<64, 256, 0, stream>>>(gates, drnn_bih, drnn_bhh, dec_h, dec_c, dech_bf);
    k_proj<<<BSZ, 128, 0, stream>>>(dec_h, ctx, proj_w, proj_b, gate_w, gate_b, out, t);
  }
}
